// SimpleSelfAttention_89163521065459
// MI455X (gfx1250) — matrix operand order repaired and checked by reading, not compiled, not run on hardware
//
#include <hip/hip_runtime.h>

#define BATCH 8
#define SEQ   2048
#define DIM   512
#define BM    32            // query rows per workgroup
#define BN    64            // key rows per tile
#define NTHR  256           // 8 waves (wave32)
#define NITER (SEQ / BN)    // 32 key tiles

// LDS row strides in elements (row starts 16B-aligned, +16B pad -> bank spread)
#define KSTR  (DIM + 8)     // bf16: 520 elems = 1040 B
#define VTSTR (BN + 8)      // bf16: 72 elems  = 144 B   (fallback kernel only)
#define SSTR  (BN + 4)      // f32 : 68 elems  = 272 B
#define PSTR  (BN + 8)      // bf16: 72 elems  = 144 B

#define XBF_BYTES  ((size_t)BATCH * SEQ * DIM * 2)   // 16,777,216
#define WS_NEEDED  (2 * XBF_BYTES)                   // 33,554,432

typedef __attribute__((ext_vector_type(16))) __bf16 v16bf;
typedef __attribute__((ext_vector_type(16))) short  v16s;
typedef __attribute__((ext_vector_type(8)))  short  v8s;
typedef __attribute__((ext_vector_type(8)))  float  v8f;

static __device__ __forceinline__ unsigned short f2bf(float f) {
  unsigned u = __float_as_uint(f);
  u += 0x7FFFu + ((u >> 16) & 1u);          // round-to-nearest-even
  return (unsigned short)(u >> 16);
}

static __device__ __forceinline__ unsigned long long pack4(unsigned short a, unsigned short b,
                                                           unsigned short c, unsigned short d) {
  return (unsigned long long)a | ((unsigned long long)b << 16)
       | ((unsigned long long)c << 32) | ((unsigned long long)d << 48);
}

// Build a 16-element bf16 fragment from two 16B chunks (LDS or global).
static __device__ __forceinline__ v16bf frag2(const __bf16* p0, const __bf16* p1) {
  v8s lo = *(const v8s*)p0;
  v8s hi = *(const v8s*)p1;
  v16s c = __builtin_shufflevector(lo, hi, 0,1,2,3,4,5,6,7,8,9,10,11,12,13,14,15);
  return __builtin_bit_cast(v16bf, c);
}

// Low 32 bits of a flat pointer to a __shared__ object == LDS byte address.
static __device__ __forceinline__ unsigned lds_addr32(const void* p) {
  return (unsigned)(unsigned long long)p;
}

static __device__ __forceinline__ void async_b128(unsigned lds_dst, const void* gsrc) {
  asm volatile("global_load_async_to_lds_b128 %0, %1, off"
               :: "v"(lds_dst), "v"((unsigned long long)gsrc)
               : "memory");
}

static __device__ __forceinline__ void wait_async0() {
  asm volatile("s_wait_asynccnt 0" ::: "memory");
}

// ---------------------------------------------------------------------------
// Prepass: x fp32 [B][T][D]  ->  ws: bf16 [B][T][D]  and  bf16 [B][D][T]
// ---------------------------------------------------------------------------
__global__ __launch_bounds__(256, 1)
void prep_bf16(const float* __restrict__ x,
               unsigned short* __restrict__ xbf,
               unsigned short* __restrict__ xbfT) {
  __shared__ unsigned short t16[64][72];     // +8 pad elems -> 144B stride

  const int tid = threadIdx.x;
  const int bid = blockIdx.x;                // 8 * 32 * 8 = 2048 blocks
  const int b   = bid >> 8;
  const int tt  = (bid >> 3) & 31;
  const int dt  = bid & 7;
  const int t0  = tt * 64, d0 = dt * 64;

  const float*    xb  = x    + ((size_t)b * SEQ + t0) * DIM + d0;
  unsigned short* ob  = xbf  + ((size_t)b * SEQ + t0) * DIM + d0;
  unsigned short* otb = xbfT + ((size_t)b * DIM + d0) * SEQ + t0;

  #pragma unroll
  for (int j = 0; j < 4; ++j) {
    const int e = j * 1024 + tid * 4;
    const int r = e >> 6;                    // t row in tile
    const int c = e & 63;                    // d col in tile (mult of 4)
    float4 v = *(const float4*)(xb + (size_t)r * DIM + c);
    unsigned short e0 = f2bf(v.x), e1 = f2bf(v.y), e2 = f2bf(v.z), e3 = f2bf(v.w);
    *(unsigned long long*)(ob + (size_t)r * DIM + c) = pack4(e0, e1, e2, e3);
    t16[r][c + 0] = e0; t16[r][c + 1] = e1; t16[r][c + 2] = e2; t16[r][c + 3] = e3;
  }
  __syncthreads();
  #pragma unroll
  for (int j = 0; j < 4; ++j) {
    const int e  = j * 1024 + tid * 4;
    const int dr = e >> 6;                   // d row of output
    const int tc = e & 63;                   // t col (mult of 4)
    *(unsigned long long*)(otb + (size_t)dr * SEQ + tc) =
        pack4(t16[tc + 0][dr], t16[tc + 1][dr], t16[tc + 2][dr], t16[tc + 3][dr]);
  }
}

// ---------------------------------------------------------------------------
// Main flash-attention kernel: Q + K tiles in LDS (async-staged),
// P*V B-operands streamed from transposed bf16 global through L1/L2.
// LDS = 113.5 KB -> two workgroups co-resident per 320KB WGP.
// ---------------------------------------------------------------------------
__global__ __launch_bounds__(NTHR, 1)
void flash_attn_ws(const unsigned short* __restrict__ xbf,
                   const unsigned short* __restrict__ xbfT,
                   float* __restrict__ y) {
  __shared__ __bf16 lds_q [BM * KSTR];      // 33,280 B
  __shared__ __bf16 lds_k [BN * KSTR];      // 66,560 B
  __shared__ float  lds_s [BM * SSTR];      //  8,704 B
  __shared__ __bf16 lds_p [BM * PSTR];      //  4,608 B
  __shared__ float  lds_m [BM];
  __shared__ float  lds_l [BM];
  __shared__ float  lds_a [BM];

  const int tid  = threadIdx.x;
  const int lane = tid & 31;
  const int wave = tid >> 5;
  const int wm   = wave >> 2;
  const int wn   = wave & 3;
  const int hlf  = lane >> 4;
  const int ln16 = lane & 15;

  const int bid = blockIdx.x;
  const int b   = bid >> 6;
  const int q0  = (bid & 63) * BM;

  const unsigned short* xbf_b  = xbf  + (size_t)b * SEQ * DIM;   // [t][d]
  const unsigned short* xbfT_b = xbfT + (size_t)b * DIM * SEQ;   // [d][t]

  v8f oacc[8] = {};
  if (tid < BM) { lds_m[tid] = -3.0e38f; lds_l[tid] = 0.0f; }

  // ---- async-stage Q tile (once) and K tile 0 ----
  {
    const char* gq = (const char*)xbf_b + (size_t)q0 * DIM * 2;
    const unsigned qb = lds_addr32(lds_q);
    #pragma unroll
    for (int i = 0; i < 8; ++i) {
      const int off = (i * NTHR + tid) * 16;             // 0..32752
      async_b128(qb + (off >> 10) * (KSTR * 2) + (off & 1023), gq + off);
    }
    const char* gk = (const char*)xbf_b;                 // k0 = 0
    const unsigned kb = lds_addr32(lds_k);
    #pragma unroll
    for (int i = 0; i < 16; ++i) {
      const int off = (i * NTHR + tid) * 16;             // 0..65520
      async_b128(kb + (off >> 10) * (KSTR * 2) + (off & 1023), gk + off);
    }
  }
  wait_async0();
  __syncthreads();

  const __bf16* aq = &lds_q[(16 * wm + ln16) * KSTR];                  // A rows (query)
  const __bf16* bk = &lds_k[(16 * wn + ln16) * KSTR + 16 * hlf];       // B rows (key)
  const __bf16* gv = (const __bf16*)(xbfT_b + (size_t)(128 * wn + ln16) * SEQ) + 16 * hlf;

  for (int kt = 0; kt < NITER; ++kt) {
    const int k0 = kt * BN;

    // warm L2/L1 for next tile while computing on this one
    if (kt + 1 < NITER) {
      const char* nk = (const char*)xbf_b + (size_t)(k0 + BN) * DIM * 2;
      __builtin_prefetch(nk + tid * 256, 0, 1);
      __builtin_prefetch(nk + tid * 256 + 128, 0, 1);
      const char* nv = (const char*)xbfT_b + (size_t)(k0 + BN) * 2;
      __builtin_prefetch(nv + (size_t)(2 * tid)     * (SEQ * 2), 0, 1);
      __builtin_prefetch(nv + (size_t)(2 * tid + 1) * (SEQ * 2), 0, 1);
    }

    // ---- S tile = Q K^T : 16 WMMA, depth-2 pipelined A and B streams ----
    v8f sacc = {};
    {
      auto aload = [&](int f) -> v16bf {       // A-layout: 16 contiguous
        const __bf16* p = aq + 32 * f + 16 * hlf;
        return frag2(p, p + 8);
      };
      auto bload = [&](int f) -> v16bf {       // B-layout: 16 contiguous
        const __bf16* p = bk + 32 * f;
        return frag2(p, p + 8);
      };
      v16bf a0 = aload(0), b0 = bload(0);
      v16bf a1 = aload(1), b1 = bload(1);
      #pragma unroll
      for (int f = 0; f < 16; ++f) {
        sacc = __builtin_amdgcn_wmma_f32_16x16x32_bf16(
            false, (f & 1) ? a1 : a0, false, (f & 1) ? b1 : b0,
            (short)0, sacc, false, false);
        if (f + 2 < 16) {
          if (f & 1) { a1 = aload(f + 2); b1 = bload(f + 2); }
          else       { a0 = aload(f + 2); b0 = bload(f + 2); }
        }
      }
    }
    {
      float* srow = &lds_s[(16 * wm + 8 * hlf) * SSTR + 16 * wn + ln16];
      #pragma unroll
      for (int r = 0; r < 8; ++r) srow[r * SSTR] = sacc[r];
    }
    __syncthreads();

    // ---- online softmax: one thread per query row ----
    if (tid < BM) {
      const float L2E = 1.44269504088896341f;
      const float* srow = &lds_s[tid * SSTR];
      float mx = -3.0e38f;
      #pragma unroll
      for (int j = 0; j < BN; ++j) mx = fmaxf(mx, srow[j]);
      float mold  = lds_m[tid];
      float mnew  = fmaxf(mold, mx);
      float alpha = exp2f((mold - mnew) * L2E);
      float lsum  = 0.f;
      unsigned* prow = (unsigned*)&((unsigned short*)lds_p)[tid * PSTR];
      #pragma unroll
      for (int j = 0; j < BN; j += 2) {
        float p0 = exp2f((srow[j]     - mnew) * L2E);
        float p1 = exp2f((srow[j + 1] - mnew) * L2E);
        lsum += p0 + p1;
        prow[j >> 1] = (unsigned)f2bf(p0) | ((unsigned)f2bf(p1) << 16);
      }
      lds_l[tid] = lds_l[tid] * alpha + lsum;
      lds_m[tid] = mnew;
      lds_a[tid] = alpha;
    }
    __syncthreads();

    // ---- rescale O, then O += P V : 16 WMMA, depth-4 pipelined global V ----
    {
      float al[8];
      #pragma unroll
      for (int r = 0; r < 8; ++r) al[r] = lds_a[16 * wm + 8 * hlf + r];
      #pragma unroll
      for (int t = 0; t < 8; ++t) {
        #pragma unroll
        for (int r = 0; r < 8; ++r) oacc[t][r] *= al[r];
      }

      const __bf16* pbase = &lds_p[(16 * wm + ln16) * PSTR];
      v16bf pa[2];
      #pragma unroll
      for (int g = 0; g < 2; ++g)
        pa[g] = frag2(pbase + 32 * g + 16 * hlf, pbase + 32 * g + 8 + 16 * hlf);

      const __bf16* vbase = gv + k0;          // global, this key window
      auto vload = [&](int s) -> v16bf {      // s = t*2 + g
        const __bf16* p = vbase + (size_t)(16 * (s >> 1)) * SEQ + 32 * (s & 1);
        return frag2(p, p + 8);
      };
      v16bf vf[4];
      #pragma unroll
      for (int j = 0; j < 4; ++j) vf[j] = vload(j);
      #pragma unroll
      for (int s = 0; s < 16; ++s) {
        oacc[s >> 1] = __builtin_amdgcn_wmma_f32_16x16x32_bf16(
            false, pa[s & 1], false, vf[s & 3], (short)0, oacc[s >> 1], false, false);
        if (s + 4 < 16) vf[s & 3] = vload(s + 4);
      }
    }
    __syncthreads();   // all waves done with lds_k / lds_p

    // ---- async-stage next K tile ----
    if (kt + 1 < NITER) {
      const char* gk = (const char*)xbf_b + (size_t)(k0 + BN) * DIM * 2;
      const unsigned kb = lds_addr32(lds_k);
      #pragma unroll
      for (int i = 0; i < 16; ++i) {
        const int off = (i * NTHR + tid) * 16;
        async_b128(kb + (off >> 10) * (KSTR * 2) + (off & 1023), gk + off);
      }
      wait_async0();
    }
    __syncthreads();
  }

  // ---- epilogue ----
  {
    float linv[8];
    #pragma unroll
    for (int r = 0; r < 8; ++r) linv[r] = 1.0f / lds_l[16 * wm + 8 * hlf + r];
    float* yb = y + (size_t)b * SEQ * DIM;
    #pragma unroll
    for (int t = 0; t < 8; ++t) {
      #pragma unroll
      for (int r = 0; r < 8; ++r) {
        const int row = q0 + 16 * wm + 8 * hlf + r;
        const int col = 128 * wn + 16 * t + ln16;
        yb[(size_t)row * DIM + col] = oacc[t][r] * linv[r];
      }
    }
  }
}

// ---------------------------------------------------------------------------
// Fallback: self-converting, used only if ws is too small.
// ---------------------------------------------------------------------------
__global__ __launch_bounds__(NTHR, 1)
void flash_attn_fb(const float* __restrict__ x, float* __restrict__ y) {
  __shared__ __bf16 lds_k [BN * KSTR];
  __shared__ __bf16 lds_vt[DIM * VTSTR];
  __shared__ float  lds_s [BM * SSTR];
  __shared__ __bf16 lds_p [BM * PSTR];
  __shared__ float  lds_m [BM];
  __shared__ float  lds_l [BM];
  __shared__ float  lds_a [BM];

  const int tid  = threadIdx.x;
  const int lane = tid & 31;
  const int wave = tid >> 5;
  const int wm   = wave >> 2;
  const int wn   = wave & 3;
  const int hlf  = lane >> 4;
  const int ln16 = lane & 15;

  const int bid = blockIdx.x;
  const int b   = bid >> 6;
  const int q0  = (bid & 63) * BM;

  const float* xb = x + (size_t)b * SEQ * DIM;

  v16bf qf[16];
  {
    const float* xq = xb + (size_t)(q0 + 16 * wm + ln16) * DIM;
    #pragma unroll
    for (int f = 0; f < 16; ++f) {
      const int d0 = 32 * f + 16 * hlf;
      const float4* g0 = (const float4*)(xq + d0);
      const float4* g1 = (const float4*)(xq + d0 + 8);
      float4 a0 = g0[0], a1 = g0[1];
      float4 b0 = g1[0], b1 = g1[1];
      v16s r;
      r[0]=(short)f2bf(a0.x);  r[1]=(short)f2bf(a0.y);  r[2]=(short)f2bf(a0.z);  r[3]=(short)f2bf(a0.w);
      r[4]=(short)f2bf(a1.x);  r[5]=(short)f2bf(a1.y);  r[6]=(short)f2bf(a1.z);  r[7]=(short)f2bf(a1.w);
      r[8]=(short)f2bf(b0.x);  r[9]=(short)f2bf(b0.y);  r[10]=(short)f2bf(b0.z); r[11]=(short)f2bf(b0.w);
      r[12]=(short)f2bf(b1.x); r[13]=(short)f2bf(b1.y); r[14]=(short)f2bf(b1.z); r[15]=(short)f2bf(b1.w);
      qf[f] = __builtin_bit_cast(v16bf, r);
    }
  }

  v8f oacc[8] = {};
  if (tid < BM) { lds_m[tid] = -3.0e38f; lds_l[tid] = 0.0f; }
  __syncthreads();

  for (int kt = 0; kt < NITER; ++kt) {
    const int k0 = kt * BN;
    {
      const float* src = xb + (size_t)k0 * DIM;
      unsigned long long* kk = (unsigned long long*)lds_k;
      unsigned short*     vt = (unsigned short*)lds_vt;
      #pragma unroll 4
      for (int st = 0; st < 32; ++st) {
        const int idx = st * 1024 + tid * 4;
        const int r   = idx >> 9;
        const int c   = idx & 511;
        float4 v = *(const float4*)(src + idx);
        unsigned short e0 = f2bf(v.x), e1 = f2bf(v.y), e2 = f2bf(v.z), e3 = f2bf(v.w);
        kk[(r * KSTR + c) >> 2] = pack4(e0, e1, e2, e3);
        vt[(c + 0) * VTSTR + r] = e0;
        vt[(c + 1) * VTSTR + r] = e1;
        vt[(c + 2) * VTSTR + r] = e2;
        vt[(c + 3) * VTSTR + r] = e3;
      }
    }
    __syncthreads();

    v8f sacc = {};
    {
      const __bf16* kbase = &lds_k[(16 * wn + ln16) * KSTR + 16 * hlf];
      #pragma unroll
      for (int f = 0; f < 16; ++f) {
        const __bf16* p = kbase + 32 * f;
        v16bf kb = frag2(p, p + 8);
        sacc = __builtin_amdgcn_wmma_f32_16x16x32_bf16(
            false, qf[f], false, kb, (short)0, sacc, false, false);
      }
    }
    {
      float* srow = &lds_s[(16 * wm + 8 * hlf) * SSTR + 16 * wn + ln16];
      #pragma unroll
      for (int r = 0; r < 8; ++r) srow[r * SSTR] = sacc[r];
    }
    __syncthreads();

    if (tid < BM) {
      const float L2E = 1.44269504088896341f;
      const float* srow = &lds_s[tid * SSTR];
      float mx = -3.0e38f;
      #pragma unroll
      for (int j = 0; j < BN; ++j) mx = fmaxf(mx, srow[j]);
      float mold  = lds_m[tid];
      float mnew  = fmaxf(mold, mx);
      float alpha = exp2f((mold - mnew) * L2E);
      float lsum  = 0.f;
      unsigned* prow = (unsigned*)&((unsigned short*)lds_p)[tid * PSTR];
      #pragma unroll
      for (int j = 0; j < BN; j += 2) {
        float p0 = exp2f((srow[j]     - mnew) * L2E);
        float p1 = exp2f((srow[j + 1] - mnew) * L2E);
        lsum += p0 + p1;
        prow[j >> 1] = (unsigned)f2bf(p0) | ((unsigned)f2bf(p1) << 16);
      }
      lds_l[tid] = lds_l[tid] * alpha + lsum;
      lds_m[tid] = mnew;
      lds_a[tid] = alpha;
    }
    __syncthreads();

    {
      float al[8];
      #pragma unroll
      for (int r = 0; r < 8; ++r) al[r] = lds_a[16 * wm + 8 * hlf + r];
      #pragma unroll
      for (int t = 0; t < 8; ++t) {
        #pragma unroll
        for (int r = 0; r < 8; ++r) oacc[t][r] *= al[r];
      }
      const __bf16* pbase = &lds_p[(16 * wm + ln16) * PSTR];
      v16bf pa[2];
      #pragma unroll
      for (int g = 0; g < 2; ++g)
        pa[g] = frag2(pbase + 32 * g + 16 * hlf, pbase + 32 * g + 8 + 16 * hlf);

      const __bf16* vroot = &lds_vt[(128 * wn + ln16) * VTSTR + 16 * hlf];
      #pragma unroll
      for (int s = 0; s < 16; ++s) {
        const __bf16* p = vroot + (s >> 1) * (16 * VTSTR) + 32 * (s & 1);
        v16bf vb = frag2(p, p + 8);
        oacc[s >> 1] = __builtin_amdgcn_wmma_f32_16x16x32_bf16(
            false, pa[s & 1], false, vb, (short)0, oacc[s >> 1], false, false);
      }
    }
    __syncthreads();
  }

  {
    float linv[8];
    #pragma unroll
    for (int r = 0; r < 8; ++r) linv[r] = 1.0f / lds_l[16 * wm + 8 * hlf + r];
    float* yb = y + (size_t)b * SEQ * DIM;
    #pragma unroll
    for (int t = 0; t < 8; ++t) {
      #pragma unroll
      for (int r = 0; r < 8; ++r) {
        const int row = q0 + 16 * wm + 8 * hlf + r;
        const int col = 128 * wn + 16 * t + ln16;
        yb[(size_t)row * DIM + col] = oacc[t][r] * linv[r];
      }
    }
  }
}

extern "C" void kernel_launch(void* const* d_in, const int* in_sizes, int n_in,
                              void* d_out, int out_size, void* d_ws, size_t ws_size,
                              hipStream_t stream) {
  (void)in_sizes; (void)n_in; (void)out_size;
  const float* x = (const float*)d_in[0];
  float* yo = (float*)d_out;

  if (ws_size >= WS_NEEDED) {
    unsigned short* xbf  = (unsigned short*)d_ws;
    unsigned short* xbfT = (unsigned short*)((char*)d_ws + XBF_BYTES);
    hipLaunchKernelGGL(prep_bf16, dim3(2048), dim3(256), 0, stream, x, xbf, xbfT);
    hipLaunchKernelGGL(flash_attn_ws, dim3(BATCH * (SEQ / BM)), dim3(NTHR), 0, stream,
                       xbf, xbfT, yo);
  } else {
    hipLaunchKernelGGL(flash_attn_fb, dim3(BATCH * (SEQ / BM)), dim3(NTHR), 0, stream, x, yo);
  }
}
